// DiGSNetwork_901943132242
// MI455X (gfx1250) — matrix operand order repaired and checked by reading, not compiled, not run on hardware
//
#include <hip/hip_runtime.h>

typedef __attribute__((ext_vector_type(16))) _Float16 v16h;
typedef __attribute__((ext_vector_type(2)))  __fp16   v2fp;
typedef __attribute__((ext_vector_type(8)))  float    v8f;

#define HID 256
#define KTOT 2048           // 8*256  (d = 1..8; d=0 folded into bias)
#define KCH_TOT 64          // 2048/32 global k-chunks
#define NSUB 4              // K processed in 4 sub-chunks of 512 (degree pairs)
#define KCH_SUB 16          // 512/32
#define KSUB 512            // phi columns per sub-chunk
#define MT 64               // points per block
#define NT 512              // threads per block (16 waves; wave w owns n-chunk w)

// ---------------- weight repack: coeff[i,o,d] (f32) -> B-fragment f16 layout ----
// k = (d-1)*I + i for d=1..8 (d=0 slice becomes the bias table).
// B-fragment (32x16 KxN) lane map: lanes 0-15 hold K rows 0..15 of column n=lane
// (VGPR v = rows 2v,2v+1), lanes 16-31 hold K rows 16..31 of column n=lane-16.
// Storage: [frag = kc*nchunks+nc][lane][2 x uint4]
__global__ __launch_bounds__(256) void kan_repack(const float* __restrict__ w,
                                                  uint4* __restrict__ dst,
                                                  int I, int O, int kchunks, int nchunks) {
    int t = blockIdx.x * 256 + threadIdx.x;
    int total = kchunks * nchunks * 32;
    if (t >= total) return;
    int lane = t & 31;
    int frag = t >> 5;
    int nc = frag % nchunks;
    int kc = frag / nchunks;
    int n  = lane & 15;
    int hi = (lane >> 4) * 16;
    int o  = nc * 16 + n;
    int K  = 8 * I;
    unsigned int dw[8];
#pragma unroll
    for (int v = 0; v < 8; ++v) {
        float f0 = 0.f, f1 = 0.f;
        int k0 = kc * 32 + hi + 2 * v;
        if (o < O) {
            if (k0 < K)     { int i = k0 % I, d = k0 / I + 1;               f0 = w[((size_t)i * O + o) * 9 + d]; }
            if (k0 + 1 < K) { int k = k0 + 1; int i = k % I, d = k / I + 1; f1 = w[((size_t)i * O + o) * 9 + d]; }
        }
        union { _Float16 h[2]; unsigned int u; } pk;
        pk.h[0] = (_Float16)f0; pk.h[1] = (_Float16)f1;
        dw[v] = pk.u;
    }
    size_t base = (size_t)t * 2;
    dst[base + 0] = make_uint4(dw[0], dw[1], dw[2], dw[3]);
    dst[base + 1] = make_uint4(dw[4], dw[5], dw[6], dw[7]);
}

// bias[o] = sum_i coeff[i][o][0]   (T_0 == 1 contribution)
__global__ __launch_bounds__(256) void kan_bias(const float* __restrict__ w,
                                                float* __restrict__ dst, int I, int O) {
    int o = threadIdx.x;
    float s = 0.f;
    if (o < O)
        for (int i = 0; i < I; ++i) s += w[((size_t)i * O + o) * 9 + 0];
    dst[o] = s;
}

// ---------------- fragment loaders ----------------
static __device__ __forceinline__ v16h load_a(const _Float16* phiB, int row, int kcl, int laneHi8) {
    const _Float16* p = phiB + (size_t)row * KSUB + kcl * 32 + 2 * laneHi8;
    union { v16h v; float4 f[2]; } u;
    u.f[0] = *(const float4*)(p);
    u.f[1] = *(const float4*)(p + 8);
    return u.v;
}
static __device__ __forceinline__ v16h load_b(const uint4* __restrict__ wf, size_t idx) {
    union { v16h v; uint4 q[2]; } u;
    u.q[0] = wf[idx];
    u.q[1] = wf[idx + 1];
    return u.v;
}
static __device__ __forceinline__ unsigned int pk2(float a, float b) {
    union { v2fp h; unsigned int u; } x;
    x.h = __builtin_amdgcn_cvt_pkrtz(a, b);   // one v_cvt_pk_rtz_f16_f32
    return x.u;
}

// phi sub-chunk: T_{2s+1}, T_{2s+2} of x (already tanh'd, in h) for all 64 rows.
// phi[p][dd*256 + i], dd in {0,1}.
static __device__ __forceinline__ void phi_sub(const float (*h)[HID],
                                               unsigned int* __restrict__ phiW,
                                               int tid, int sub) {
    const int dmax = 2 * sub + 2;
#pragma unroll 4
    for (int t = 0; t < 16; ++t) {
        int item = tid + t * NT;              // 0..8191
        int p  = item >> 7;
        int i2 = (item & 127) << 1;
        const float* hp = &h[p][i2];
        float xa = hp[0], xb = hp[1];
        float tpa = 1.f, tca = xa, tpb = 1.f, tcb = xb;
        for (int d = 2; d <= dmax; ++d) {
            float tna = 2.f * xa * tca - tpa;
            float tnb = 2.f * xb * tcb - tpb;
            tpa = tca; tca = tna;
            tpb = tcb; tcb = tnb;
        }
        // after loop: tp = T_{2s+1}, tc = T_{2s+2}
        size_t base = ((size_t)p * KSUB + i2) >> 1;    // dword index
        phiW[base]       = pk2(tpa, tpb);
        phiW[base + 128] = pk2(tca, tcb);
    }
}

// ---------------- fused network kernel ----------------
// One block = 64 points through all 8 layers, 16 waves (one n-chunk each).
// LDS: h/x[64][256] f32 (64KB, tanh in place) + phi[64][512] f16 (64KB) = 128KB.
__global__ __launch_bounds__(NT) void kan_fused(const float* __restrict__ mnfld,
                                                const float* __restrict__ nonm,
                                                const uint4* __restrict__ wfrag,
                                                const float* __restrict__ biasTab,
                                                float* __restrict__ out) {
    extern __shared__ char smem_raw[];
    float (*h)[HID]    = (float (*)[HID])smem_raw;
    _Float16* phiB     = (_Float16*)(smem_raw + MT * HID * sizeof(float));
    unsigned int* phiW = (unsigned int*)phiB;

    const int tid  = threadIdx.x;
    const int lane = tid & 31;
    const int wave = tid >> 5;            // 0..15 == n-chunk
    const int laneHi8 = (lane & 16) ? 8 : 0;
    const int row   = lane & 15;
    const int mbase = (lane & 16) ? 8 : 0;
    const int col   = wave * 16 + row;

    int b = blockIdx.x;
    const float* src;
    size_t outOff;
    if (b < 1024) { src = mnfld + (size_t)b * MT * 3;          outOff = (size_t)b * MT; }
    else          { src = nonm  + (size_t)(b - 1024) * MT * 3; outOff = 65536 + (size_t)(b - 1024) * MT; }

    const size_t OFF_MID = 1024;                        // after L0 (16 frags * 64 uint4)
    const size_t FR_MID  = (size_t)KCH_TOT * 16 * 64;   // 65536 uint4 per middle layer
    const size_t OFF_L7  = OFF_MID + 6 * FR_MID;        // 394240

    // ---- layer 0: phi from raw input (K = 24 padded to 32, single k-chunk) ----
    if (tid < MT) {
        int p = tid;
        float x[3];
#pragma unroll
        for (int i = 0; i < 3; ++i) x[i] = tanhf(src[p * 3 + i]);
        _Float16* prow = phiB + (size_t)p * KSUB;
#pragma unroll
        for (int i = 0; i < 3; ++i) {
            float tp = 1.f, tc = x[i];
            prow[i] = (_Float16)tc;                    // d=1
#pragma unroll
            for (int d = 2; d <= 8; ++d) {
                float tn = 2.f * x[i] * tc - tp;
                prow[(d - 1) * 3 + i] = (_Float16)tn;
                tp = tc; tc = tn;
            }
        }
#pragma unroll
        for (int k = 24; k < 32; ++k) prow[k] = (_Float16)0.f;
    }
    __syncthreads();
    {
        float bv = biasTab[col];
        v16h bm = load_b(wfrag, (size_t)wave * 64 + lane * 2);
#pragma unroll
        for (int mt = 0; mt < 4; ++mt) {
            v8f c;
#pragma unroll
            for (int r = 0; r < 8; ++r) c[r] = bv;
            v16h a = load_a(phiB, mt * 16 + row, 0, laneHi8);
            c = __builtin_amdgcn_wmma_f32_16x16x32_f16(false, a, false, bm, (short)0, c, false, false);
#pragma unroll
            for (int r = 0; r < 8; ++r) h[mt * 16 + mbase + r][col] = c[r];
        }
    }
    __syncthreads();

    // ---- layers 1..6 ----
    for (int l = 1; l <= 6; ++l) {
        // x = tanh(h) in place, float4 quads
#pragma unroll 4
        for (int t = 0; t < 8; ++t) {
            int idx = tid + t * NT;                   // 4096 quads
            float4* q = (float4*)smem_raw + idx;
            float4 v = *q;
            v.x = tanhf(v.x); v.y = tanhf(v.y); v.z = tanhf(v.z); v.w = tanhf(v.w);
            *q = v;
        }
        __syncthreads();

        const uint4* wf = wfrag + OFF_MID + (size_t)(l - 1) * FR_MID;
        float bv = biasTab[(size_t)l * 256 + col];
        v8f c[4];
#pragma unroll
        for (int mt = 0; mt < 4; ++mt)
#pragma unroll
            for (int r = 0; r < 8; ++r) c[mt][r] = bv;

#pragma unroll 1
        for (int sub = 0; sub < NSUB; ++sub) {
            phi_sub(h, phiW, tid, sub);
            __syncthreads();
#pragma unroll 2
            for (int kcl = 0; kcl < KCH_SUB; ++kcl) {
                int kg = sub * KCH_SUB + kcl;
                v16h bm = load_b(wf, ((size_t)(kg * 16 + wave)) * 64 + lane * 2);
#pragma unroll
                for (int mt = 0; mt < 4; ++mt) {
                    v16h a = load_a(phiB, mt * 16 + row, kcl, laneHi8);
                    c[mt] = __builtin_amdgcn_wmma_f32_16x16x32_f16(false, a, false, bm,
                                                                   (short)0, c[mt], false, false);
                }
            }
            __syncthreads();   // phi is rewritten next sub-chunk
        }
        // write next-layer activations
#pragma unroll
        for (int mt = 0; mt < 4; ++mt)
#pragma unroll
            for (int r = 0; r < 8; ++r)
                h[mt * 16 + mbase + r][col] = c[mt][r];
        __syncthreads();
    }

    // ---- layer 7: N=1 (padded to 16) ----
#pragma unroll 4
    for (int t = 0; t < 8; ++t) {
        int idx = tid + t * NT;
        float4* q = (float4*)smem_raw + idx;
        float4 v = *q;
        v.x = tanhf(v.x); v.y = tanhf(v.y); v.z = tanhf(v.z); v.w = tanhf(v.w);
        *q = v;
    }
    __syncthreads();
    {
        v8f c[4];
        float bv = biasTab[7 * 256 + row];     // zero for padded columns
#pragma unroll
        for (int mt = 0; mt < 4; ++mt)
#pragma unroll
            for (int r = 0; r < 8; ++r) c[mt][r] = bv;
#pragma unroll 1
        for (int sub = 0; sub < NSUB; ++sub) {
            phi_sub(h, phiW, tid, sub);
            __syncthreads();
            if (wave == 0) {
#pragma unroll 2
                for (int kcl = 0; kcl < KCH_SUB; ++kcl) {
                    int kg = sub * KCH_SUB + kcl;
                    v16h bm = load_b(wfrag, OFF_L7 + (size_t)kg * 64 + lane * 2);
#pragma unroll
                    for (int mt = 0; mt < 4; ++mt) {
                        v16h a = load_a(phiB, mt * 16 + row, kcl, laneHi8);
                        c[mt] = __builtin_amdgcn_wmma_f32_16x16x32_f16(false, a, false, bm,
                                                                       (short)0, c[mt], false, false);
                    }
                }
            }
            __syncthreads();
        }
        if (wave == 0 && row == 0) {           // column N=0: lane 0 -> M 0..7, lane 16 -> M 8..15
#pragma unroll
            for (int mt = 0; mt < 4; ++mt) {
                float* o = out + outOff + mt * 16 + ((lane & 16) ? 8 : 0);
#pragma unroll
                for (int r = 0; r < 8; ++r) o[r] = c[mt][r];
            }
        }
    }
}

extern "C" void kernel_launch(void* const* d_in, const int* in_sizes, int n_in,
                              void* d_out, int out_size, void* d_ws, size_t ws_size,
                              hipStream_t stream) {
    const float* nonm  = (const float*)d_in[0];   // non_mnfld_pnts
    const float* mnfld = (const float*)d_in[1];   // mnfld_pnts
    const float* w[8];
    for (int i = 0; i < 8; ++i) w[i] = (const float*)d_in[2 + i];

    uint4* wf = (uint4*)d_ws;
    const size_t OFF_MID = 1024;
    const size_t FR_MID  = (size_t)KCH_TOT * 16 * 64;      // 65536
    const size_t OFF_L7  = OFF_MID + 6 * FR_MID;           // 394240
    const size_t OFF_END = OFF_L7 + (size_t)KCH_TOT * 64;  // 398336 uint4
    float* biasTab = (float*)((char*)d_ws + OFF_END * sizeof(uint4));

    kan_repack<<<2, 256, 0, stream>>>(w[0], wf, 3, 256, 1, 16);
    for (int l = 1; l <= 6; ++l)
        kan_repack<<<128, 256, 0, stream>>>(w[l], wf + OFF_MID + (size_t)(l - 1) * FR_MID,
                                            256, 256, KCH_TOT, 16);
    kan_repack<<<8, 256, 0, stream>>>(w[7], wf + OFF_L7, 256, 1, KCH_TOT, 1);

    kan_bias<<<1, 256, 0, stream>>>(w[0], biasTab + 0 * 256, 3, 256);
    for (int l = 1; l <= 6; ++l)
        kan_bias<<<1, 256, 0, stream>>>(w[l], biasTab + (size_t)l * 256, 256, 256);
    kan_bias<<<1, 256, 0, stream>>>(w[7], biasTab + 7 * 256, 256, 1);

    const size_t smem = MT * HID * sizeof(float) + (size_t)MT * KSUB * sizeof(_Float16); // 131072 B
    (void)hipFuncSetAttribute((const void*)kan_fused, hipFuncAttributeMaxDynamicSharedMemorySize, (int)smem);
    kan_fused<<<2048, NT, smem, stream>>>(mnfld, nonm, wf, biasTab, (float*)d_out);
}
